// RMPI_7550552506743
// MI455X (gfx1250) — matrix operand order repaired and checked by reading, not compiled, not run on hardware
//
#include <hip/hip_runtime.h>
#include <hip/hip_bf16.h>

typedef __attribute__((ext_vector_type(16))) _Float16 v16h;
typedef __attribute__((ext_vector_type(8)))  float    v8f;

#define E_EDGES 16384
#define B_ROWS  2048
#define NUM_REL 200
#define TILES   (E_EDGES / 16)
#define WAVES   8

// out[e][d] = rel_emb[edge_type[e]][d]  (h0 pass-through for all E edges)
__global__ __launch_bounds__(256) void rmpi_copy_h0(const int* __restrict__ edge_type,
                                                    const float* __restrict__ rel_emb,
                                                    float* __restrict__ out) {
    int idx = blockIdx.x * 256 + threadIdx.x;     // over E*32
    int e = idx >> 5;
    int d = idx & 31;
    out[idx] = rel_emb[edge_type[e] * 32 + d];
}

// One block = 16 target rows. rel_emb lives in LDS as f16 (B-fragment-ready
// contiguous rows); dense WMMA scoring over all E edges; sparse softmax-
// weighted accumulation of h0 into LDS atomics; factored 6x(32x32) matvec +
// bias + relu; overwrite of the 16 neighbor rows of `out`.
__global__ __launch_bounds__(256) void rmpi_main(const int* __restrict__ edge_src,
                                                 const int* __restrict__ edge_dst,
                                                 const int* __restrict__ edge_type,
                                                 const int* __restrict__ nbr_edges,
                                                 const int* __restrict__ nbr_rels,
                                                 const float* __restrict__ rel_emb,
                                                 const float* __restrict__ W1,
                                                 const float* __restrict__ b1,
                                                 float* __restrict__ out) {
    __shared__ _Float16 relH[NUM_REL * 32];  // f16 copy of rel_emb (12.8 KB)
    __shared__ float numer[16][6][32];       // sum_e w_e * h0[e][:]
    __shared__ float denom[16][6];           // sum_e w_e

    const int tid  = threadIdx.x;
    const int wg   = blockIdx.x;
    const int lane = tid & 31;
    const int wave = tid >> 5;
    const int n    = lane & 15;          // A row M / B column N / D column N
    const int hh   = lane >> 4;          // lane half

    // convert rel_emb -> f16 LDS table; zero accumulators
    for (int i = tid; i < NUM_REL * 32; i += 256) relH[i] = (_Float16)rel_emb[i];
    for (int i = tid; i < 16 * 6 * 32; i += 256) ((float*)numer)[i] = 0.f;
    if (tid < 96) ((float*)denom)[tid] = 0.f;

    // ---- A fragment: xxx = rel_emb[nbr_rels[b]] for row M = lane%16 ----
    v16h afrag;
    {
        const float* xr = rel_emb + nbr_rels[wg * 16 + n] * 32;
        #pragma unroll
        for (int r = 0; r < 8; ++r) {
            int k0 = 16 * hh + 2 * r;
            afrag[2 * r]     = (_Float16)xr[k0];
            afrag[2 * r + 1] = (_Float16)xr[k0 + 1];
        }
    }

    // ---- per-lane u,v for the D-matrix rows m = r + 8*hh ----
    int u8[8], v8[8];
    #pragma unroll
    for (int r = 0; r < 8; ++r) {
        int ne = nbr_edges[wg * 16 + r + 8 * hh];
        u8[r] = edge_src[ne];
        v8[r] = edge_dst[ne];
    }
    __syncthreads();

    // ---- dense scoring over all E edges, 16-edge tiles per wave ----
    for (int tile = wave; tile < TILES; tile += WAVES) {
        int e  = tile * 16 + n;
        int t  = edge_type[e];
        int es = edge_src[e];
        int ed = edge_dst[e];

        // B fragment: B[k][n] = h0[e_n][k]; lane holds K = 16*hh + j,
        // i.e. 32 contiguous bytes of the LDS f16 row (32B aligned).
        v16h bfrag = *(const v16h*)(relH + t * 32 + 16 * hh);

        v8f c = {0.f, 0.f, 0.f, 0.f, 0.f, 0.f, 0.f, 0.f};
        c = __builtin_amdgcn_wmma_f32_16x16x32_f16(false, afrag, false, bfrag,
                                                   (short)0, c, false, false);

        #pragma unroll
        for (int r = 0; r < 8; ++r) {
            float sraw = c[r];                    // pre-leaky_relu score
            bool in_u  = (ed == u8[r]);
            bool out_u = (es == u8[r]);
            bool in_v  = (ed == v8[r]);
            bool out_v = (es == v8[r]);
            if (sraw != 0.f && (in_u | out_u | in_v | out_v)) {
                bool m5 = out_u && in_v;
                bool m6 = in_u && out_v;
                unsigned mb = (unsigned)(in_u && !m6)
                            | ((unsigned)(out_u && !m5) << 1)
                            | ((unsigned)(in_v && !m5) << 2)
                            | ((unsigned)(out_v && !m6) << 3)
                            | ((unsigned)m5 << 4)
                            | ((unsigned)m6 << 5);
                float s   = sraw > 0.f ? sraw : 0.2f * sraw;   // leaky_relu
                float wgt = __expf(s);                          // no max-shift needed
                int   m   = r + 8 * hh;
                const float* hr = rel_emb + t * 32;             // f32 h0[e][:]
                #pragma unroll
                for (int i = 0; i < 6; ++i) {
                    if ((mb >> i) & 1u) {
                        atomicAdd(&denom[m][i], wgt);
                        for (int k = 0; k < 32; ++k)
                            atomicAdd(&numer[m][i][k], wgt * hr[k]);
                    }
                }
            }
        }
    }
    __syncthreads();

    // ---- finalize: agg = sum_i ( W1[i] @ (numer/denom) + b1[i]*has_i ) ----
    int m = tid >> 4;
    #pragma unroll
    for (int dd = 0; dd < 2; ++dd) {
        int d = ((tid & 15) << 1) + dd;
        float acc = 0.f;
        #pragma unroll
        for (int i = 0; i < 6; ++i) {
            float den = denom[m][i];
            if (den > 0.f) {
                const float* wrow = W1 + (i * 32 + d) * 32;    // W1[i][d][:]
                float dot = 0.f;
                for (int k = 0; k < 32; ++k) dot += wrow[k] * numer[m][i][k];
                acc += dot / den + b1[i * 32 + d];
            }
        }
        float relu = acc > 0.f ? acc : 0.f;
        int ne = nbr_edges[wg * 16 + m];
        out[ne * 32 + d] = rel_emb[edge_type[ne] * 32 + d] + relu;
    }
}

extern "C" void kernel_launch(void* const* d_in, const int* in_sizes, int n_in,
                              void* d_out, int out_size, void* d_ws, size_t ws_size,
                              hipStream_t stream) {
    const int*   edge_src  = (const int*)d_in[0];
    const int*   edge_dst  = (const int*)d_in[1];
    const int*   edge_type = (const int*)d_in[2];
    const int*   nbr_edges = (const int*)d_in[3];
    const int*   nbr_rels  = (const int*)d_in[4];
    const float* rel_emb   = (const float*)d_in[5];
    const float* W1        = (const float*)d_in[6];
    const float* b1        = (const float*)d_in[7];
    float* out = (float*)d_out;

    rmpi_copy_h0<<<(E_EDGES * 32) / 256, 256, 0, stream>>>(edge_type, rel_emb, out);
    rmpi_main<<<B_ROWS / 16, 256, 0, stream>>>(edge_src, edge_dst, edge_type,
                                               nbr_edges, nbr_rels, rel_emb, W1, b1, out);
}
